// MultiHeadAttention_54597624267430
// MI455X (gfx1250) — matrix operand order repaired and checked by reading, not compiled, not run on hardware
//
#include <hip/hip_runtime.h>
#include <hip/hip_fp16.h>

typedef __attribute__((ext_vector_type(16))) _Float16 v16h;
typedef __attribute__((ext_vector_type(8)))  _Float16 v8h;
typedef __attribute__((ext_vector_type(8)))  float    v8f;
typedef __attribute__((ext_vector_type(4)))  unsigned int v4u;
typedef __attribute__((ext_vector_type(8)))  int v8i;
typedef __attribute__((ext_vector_type(4)))  int v4i;

#define DIM   1024
#define NHEAD 16
#define HDIM  64
#define BATCH 4
#define SEQ   2048
#define ROWS  (BATCH * SEQ)   // 8192

// ---------------------------------------------------------------------------
// WMMA helpers (CDNA5 gfx1250, wave32). D = A(16x32 f16) * B(32x16 f16) + C(f32)
// ---------------------------------------------------------------------------
__device__ __forceinline__ v8f wmma16(v16h a, v16h b, v8f c) {
  return __builtin_amdgcn_wmma_f32_16x16x32_f16(false, a, false, b, (short)0, c,
                                                false, false);
}

// A-operand (16x32, MxK), row-major source, leading dim `ld` (elements).
__device__ __forceinline__ v16h load_a_frag(const _Float16* base, int row0,
                                            int k0, int ld) {
  int lane = threadIdx.x & 31;
  int r  = row0 + (lane & 15);
  int kb = k0 + ((lane >> 4) & 1) * 8;
  const _Float16* p = base + (size_t)r * ld + kb;
  union { v16h v; v8h h[2]; } u;
  u.h[0] = *reinterpret_cast<const v8h*>(p);
  u.h[1] = *reinterpret_cast<const v8h*>(p + 16);
  return u.v;
}

// B-operand (32x16, KxN) with B[k][n] = M[n][k], M row-major with leading dim
// `ld` ("multiply by M^T"). Two 16-byte loads per lane (k = 8h+s, 16+8h+s).
__device__ __forceinline__ v16h load_b_fragT(const _Float16* base, int col0,
                                             int k0, int ld) {
  int lane = threadIdx.x & 31;
  int n  = col0 + (lane & 15);
  int kb = k0 + ((lane >> 4) & 1) * 8;
  const _Float16* p = base + (size_t)n * ld + kb;
  union { v16h v; v8h h[2]; } u;
  u.h[0] = *reinterpret_cast<const v8h*>(p);
  u.h[1] = *reinterpret_cast<const v8h*>(p + 16);
  return u.v;
}

// Row reductions across a 16-lane half-wave (C rows live per half-wave).
__device__ __forceinline__ float rowmax16(float v) {
  v = fmaxf(v, __shfl_xor(v, 1, 32));
  v = fmaxf(v, __shfl_xor(v, 2, 32));
  v = fmaxf(v, __shfl_xor(v, 4, 32));
  v = fmaxf(v, __shfl_xor(v, 8, 32));
  return v;
}
__device__ __forceinline__ float rowsum16(float v) {
  v += __shfl_xor(v, 1, 32);
  v += __shfl_xor(v, 2, 32);
  v += __shfl_xor(v, 4, 32);
  v += __shfl_xor(v, 8, 32);
  return v;
}

// ---------------------------------------------------------------------------
// Tensor Data Mover: DMA a 2-D tile (rows x row_bytes, 8-byte units) from
// global memory into LDS (rows packed contiguously).  D# built per
// CDNA5 ISA; tensor_dims set huge so OOB never triggers.
// Tracked with TENSORcnt.
// ---------------------------------------------------------------------------
__device__ __forceinline__ unsigned lds_u32(const void* p) {
  // Generic pointers to LDS carry the wave-relative LDS byte offset in [31:0].
  return (unsigned)(unsigned long long)p;
}

__device__ __forceinline__ void tdm_load_2d(unsigned lds_off, const void* gsrc,
                                            unsigned tile_d0_units,
                                            unsigned tile_d1,
                                            unsigned stride_units) {
  unsigned long long ga = (unsigned long long)gsrc;
  v4u g0;
  g0[0] = 1u;                                    // count=1 (valid user D#)
  g0[1] = lds_off;                               // lds_addr (bytes)
  g0[2] = (unsigned)ga;                          // global_addr[31:0]
  g0[3] = (unsigned)(ga >> 32) | (2u << 30);     // global_addr[56:32] | type=2
  v8i g1;
  g1[0] = (int)(3u << 16);                       // data_size=8B, no multicast
  g1[1] = (int)(0xFFFFu << 16);                  // tensor_dim0 lo16 = 0xFFFF
  g1[2] = (int)0xFFFFFFFFu;                      // tensor_dim0 hi / dim1 lo
  g1[3] = (int)(0xFFFFu | (tile_d0_units << 16));// tensor_dim1 hi | tile_dim0
  g1[4] = (int)(tile_d1 & 0xFFFFu);              // tile_dim1 | tile_dim2=0
  g1[5] = (int)stride_units;                     // tensor_dim0_stride lo32
  g1[6] = 0;                                     // stride hi | dim1_stride lo
  g1[7] = 0;
  v4i z4 = {0, 0, 0, 0};
#if __has_include(<hip/amd_detail/amd_gfx1250_TDM.h>)
  v8i z8 = {0, 0, 0, 0, 0, 0, 0, 0};
  __builtin_amdgcn_tensor_load_to_lds(g0, g1, z4, z4, z8, 0);
#else
  __builtin_amdgcn_tensor_load_to_lds(g0, g1, z4, z4, 0);
#endif
}

// ---------------------------------------------------------------------------
// Kernel 1: f32 -> f16 conversion
// ---------------------------------------------------------------------------
__global__ void cvt_f32_f16(const float* __restrict__ in,
                            _Float16* __restrict__ out, int n) {
  int i = blockIdx.x * blockDim.x + threadIdx.x;
  if (i < n) out[i] = (_Float16)in[i];
}

// ---------------------------------------------------------------------------
// Kernel 2: QKV GEMM (8192x1024 @ 1024x3072) + bias + RoPE.
// Block = 256 threads = 8 waves, 2(M) x 4(N); wave tile 32x64 (8 accums,
// 2 A-frags reused over 4 B-frags -> ~11 flop/byte of L2 traffic).
// ---------------------------------------------------------------------------
__global__ void qkv_rope_kernel(const _Float16* __restrict__ Xh,
                                const _Float16* __restrict__ Wqkv,
                                const float* __restrict__ bqkv,
                                _Float16* __restrict__ Qh,
                                _Float16* __restrict__ Kh,
                                _Float16* __restrict__ Vt) {
  int wave = threadIdx.x >> 5;
  int lane = threadIdx.x & 31;
  int mw = wave & 1, nw = wave >> 1;
  int Mb = blockIdx.x * 64 + mw * 32;
  int Nb = blockIdx.y * 256 + nw * 64;

  v8f acc[2][4] = {};
  for (int k = 0; k < DIM; k += 32) {
    v16h a0 = load_a_frag(Xh, Mb, k, DIM);
    v16h a1 = load_a_frag(Xh, Mb + 16, k, DIM);
#pragma unroll
    for (int t = 0; t < 4; ++t) {
      v16h b = load_b_fragT(Wqkv, Nb + 16 * t, k, DIM);
      acc[0][t] = wmma16(a0, b, acc[0][t]);
      acc[1][t] = wmma16(a1, b, acc[1][t]);
    }
  }

  int hi = (lane >> 4) & 1;
#pragma unroll
  for (int ms = 0; ms < 2; ++ms) {
    int Mbase = Mb + ms * 16;
#pragma unroll
    for (int t = 0; t < 4; ++t) {
      int colg = Nb + 16 * t + (lane & 15);  // j/h uniform across the tile
      int j  = colg >> 10;                   // 0=q 1=k 2=v
      int d  = colg & 1023;
      int h  = d >> 6;
      int dh = d & 63;
      float bias = bqkv[colg];
      if (j < 2) {
        // RoPE: emb[dh] = pos * inv_freq[dh & 31], inv_freq[i]=10000^(-2i/64)
        float invf = __powf(10000.0f, -((float)(2 * (dh & 31))) * (1.0f / 64.0f));
        float sgn = (dh & 1) ? 1.0f : -1.0f;
        _Float16* dst = (j == 0) ? Qh : Kh;
#pragma unroll
        for (int r = 0; r < 8; ++r) {
          int row = Mbase + r + hi * 8;
          int bi  = row >> 11;
          int pos = row & 2047;
          float v  = acc[ms][t][r] + bias;
          float nb = __shfl_xor(v, 1, 32);   // value at dh^1 (neighbor lane)
          float s, c;
          __sincosf((float)pos * invf, &s, &c);
          float res = v * c + sgn * nb * s;
          dst[(((size_t)(bi * NHEAD + h) * SEQ + pos) << 6) + dh] = (_Float16)res;
        }
      } else {
#pragma unroll
        for (int r = 0; r < 8; ++r) {
          int row = Mbase + r + hi * 8;
          int bi  = row >> 11;
          int pos = row & 2047;
          float v = acc[ms][t][r] + bias;
          Vt[((size_t)(bi * NHEAD + h) * HDIM + dh) * SEQ + pos] = (_Float16)v;
        }
      }
    }
  }
}

// ---------------------------------------------------------------------------
// Kernel 3: flash attention.  8 waves/block share one (b,h): TDM stages each
// 32-key K-strip (4KB) and V^T-strip (4KB) into double-buffered LDS once per
// block; waves each own a 16-query tile.  Online softmax with half-wave
// shuffles; P transposed C->A layout through per-wave LDS.
// ---------------------------------------------------------------------------
__global__ void attn_kernel(const _Float16* __restrict__ Qh,
                            const _Float16* __restrict__ Kh,
                            const _Float16* __restrict__ Vt,
                            const int* __restrict__ mask,
                            _Float16* __restrict__ Out) {
  __shared__ alignas(128) _Float16 Ks[2][32 * HDIM];  // [key][dh]
  __shared__ alignas(128) _Float16 Vs[2][HDIM * 32];  // [dh][key]
  __shared__ alignas(64)  _Float16 Pbuf[8][16 * 32];

  int wave = threadIdx.x >> 5;
  int lane = threadIdx.x & 31;
  int bh  = blockIdx.x >> 4;            // 16 blocks per (b,h)
  int qt0 = (blockIdx.x & 15) * 8;      // 8 query tiles per block
  int qt  = qt0 + wave;
  int b = bh >> 4;
  int h = bh & 15;
  const _Float16* Qb = Qh + (size_t)bh * SEQ * HDIM;
  const _Float16* Kb = Kh + (size_t)bh * SEQ * HDIM;
  const _Float16* Vb = Vt + (size_t)bh * HDIM * SEQ;
  const int* mb = mask + b * SEQ;
  _Float16* Pw = Pbuf[wave];

  v16h qa0 = load_a_frag(Qb, qt * 16, 0, HDIM);
  v16h qa1 = load_a_frag(Qb, qt * 16, 32, HDIM);

  v8f o[4] = {};
  float mrow[8], lrow[8];
#pragma unroll
  for (int r = 0; r < 8; ++r) { mrow[r] = -1e30f; lrow[r] = 0.0f; }

  int hi  = (lane >> 4) & 1;
  int col = lane & 15;
  const float scale = 0.125f;   // 1/sqrt(64)

  // Prologue: stage strip 0 (K: 4KB contiguous; V: 64 rows x 64B, pitch 4KB).
  if (wave == 0) {
    tdm_load_2d(lds_u32(Ks[0]), Kb, 512, 1, 512);
    tdm_load_2d(lds_u32(Vs[0]), Vb, 8, HDIM, 512);
  }

  for (int it = 0; it < SEQ / 32; ++it) {
    int cur = it & 1, nxt = cur ^ 1;
    int key0 = it * 32;
    if (wave == 0) __builtin_amdgcn_s_wait_tensorcnt(0);  // strip `it` landed
    __syncthreads();  // publish buffers; all prior reads of buf[nxt] retired
    if (wave == 0 && it + 1 < SEQ / 32) {
      tdm_load_2d(lds_u32(Ks[nxt]), Kb + (size_t)(key0 + 32) * HDIM, 512, 1, 512);
      tdm_load_2d(lds_u32(Vs[nxt]), Vb + (key0 + 32), 8, HDIM, 512);
    }

    v8f s0 = {}, s1 = {};
    s0 = wmma16(qa0, load_b_fragT(Ks[cur], 0, 0, HDIM), s0);
    s0 = wmma16(qa1, load_b_fragT(Ks[cur], 0, 32, HDIM), s0);
    s1 = wmma16(qa0, load_b_fragT(Ks[cur], 16, 0, HDIM), s1);
    s1 = wmma16(qa1, load_b_fragT(Ks[cur], 16, 32, HDIM), s1);

    float mk0 = (mb[key0 + col] == 0) ? -1e30f : 0.0f;
    float mk1 = (mb[key0 + 16 + col] == 0) ? -1e30f : 0.0f;

#pragma unroll
    for (int r = 0; r < 8; ++r) {
      float a0 = s0[r] * scale + mk0;
      float a1 = s1[r] * scale + mk1;
      float tmax = rowmax16(fmaxf(a0, a1));
      float mnew = fmaxf(mrow[r], tmax);
      float alpha = __expf(mrow[r] - mnew);
      float p0 = __expf(a0 - mnew);
      float p1 = __expf(a1 - mnew);
      lrow[r] = lrow[r] * alpha + rowsum16(p0 + p1);
      mrow[r] = mnew;
      o[0][r] *= alpha; o[1][r] *= alpha; o[2][r] *= alpha; o[3][r] *= alpha;
      Pw[(r + hi * 8) * 32 + col]      = (_Float16)p0;
      Pw[(r + hi * 8) * 32 + 16 + col] = (_Float16)p1;
    }
    // Cross-lane LDS writes of P must land before A-fragment readback.
    asm volatile("s_wait_dscnt 0" ::: "memory");
    v16h pa = load_a_frag(Pw, 0, 0, 32);
#pragma unroll
    for (int t = 0; t < 4; ++t) {
      v16h vb = load_b_fragT(Vs[cur], t * 16, 0, 32);
      o[t] = wmma16(pa, vb, o[t]);
    }
    asm volatile("s_wait_dscnt 0" ::: "memory");
  }

#pragma unroll
  for (int r = 0; r < 8; ++r) {
    float inv = 1.0f / lrow[r];
    int qpos = qt * 16 + r + hi * 8;
    size_t base = ((size_t)(b * SEQ + qpos)) * DIM + (size_t)h * HDIM;
#pragma unroll
    for (int t = 0; t < 4; ++t) {
      Out[base + t * 16 + col] = (_Float16)(o[t][r] * inv);
    }
  }
}

// ---------------------------------------------------------------------------
// Kernel 4: output projection (8192x1024 @ 1024x1024) + bias -> f32.
// Same 32x64 wave tiling as the QKV GEMM.
// ---------------------------------------------------------------------------
__global__ void proj_kernel(const _Float16* __restrict__ A,
                            const _Float16* __restrict__ W,
                            const float* __restrict__ bias,
                            float* __restrict__ out) {
  int wave = threadIdx.x >> 5;
  int lane = threadIdx.x & 31;
  int mw = wave & 1, nw = wave >> 1;
  int Mb = blockIdx.x * 64 + mw * 32;
  int Nb = blockIdx.y * 256 + nw * 64;

  v8f acc[2][4] = {};
  for (int k = 0; k < DIM; k += 32) {
    v16h a0 = load_a_frag(A, Mb, k, DIM);
    v16h a1 = load_a_frag(A, Mb + 16, k, DIM);
#pragma unroll
    for (int t = 0; t < 4; ++t) {
      v16h bfr = load_b_fragT(W, Nb + 16 * t, k, DIM);
      acc[0][t] = wmma16(a0, bfr, acc[0][t]);
      acc[1][t] = wmma16(a1, bfr, acc[1][t]);
    }
  }

  int hi = (lane >> 4) & 1;
#pragma unroll
  for (int ms = 0; ms < 2; ++ms) {
#pragma unroll
    for (int t = 0; t < 4; ++t) {
      int colg = Nb + 16 * t + (lane & 15);
      float bv = bias[colg];
#pragma unroll
      for (int r = 0; r < 8; ++r) {
        int row = Mb + ms * 16 + r + hi * 8;
        out[(size_t)row * DIM + colg] = acc[ms][t][r] + bv;
      }
    }
  }
}

// ---------------------------------------------------------------------------
extern "C" void kernel_launch(void* const* d_in, const int* in_sizes, int n_in,
                              void* d_out, int out_size, void* d_ws,
                              size_t ws_size, hipStream_t stream) {
  const float* x      = (const float*)d_in[0];
  const int*   mask   = (const int*)d_in[1];
  const float* w_qkv  = (const float*)d_in[2];
  const float* b_qkv  = (const float*)d_in[3];
  const float* w_proj = (const float*)d_in[4];
  const float* b_proj = (const float*)d_in[5];
  float* out = (float*)d_out;

  // Workspace carve-up (all f16): ~88 MB total
  _Float16* Xh     = (_Float16*)d_ws;                  // 8192*1024
  _Float16* Wqkvh  = Xh + (size_t)ROWS * DIM;          // 3072*1024
  _Float16* Wprojh = Wqkvh + (size_t)3 * DIM * DIM;    // 1024*1024
  _Float16* Qh     = Wprojh + (size_t)DIM * DIM;       // (B,H,N,Dh)
  _Float16* Kh     = Qh + (size_t)ROWS * DIM;          // (B,H,N,Dh)
  _Float16* Vt     = Kh + (size_t)ROWS * DIM;          // (B,H,Dh,N)
  _Float16* Ao     = Vt + (size_t)ROWS * DIM;          // (B,N,D) attn out

  int nx = ROWS * DIM;
  cvt_f32_f16<<<(nx + 255) / 256, 256, 0, stream>>>(x, Xh, nx);
  int nw = 3 * DIM * DIM;
  cvt_f32_f16<<<(nw + 255) / 256, 256, 0, stream>>>(w_qkv, Wqkvh, nw);
  int np = DIM * DIM;
  cvt_f32_f16<<<(np + 255) / 256, 256, 0, stream>>>(w_proj, Wprojh, np);

  qkv_rope_kernel<<<dim3(ROWS / 64, (3 * DIM) / 256), 256, 0, stream>>>(
      Xh, Wqkvh, b_qkv, Qh, Kh, Vt);

  attn_kernel<<<BATCH * NHEAD * 16, 256, 0, stream>>>(Qh, Kh, Vt, mask, Ao);

  proj_kernel<<<dim3(ROWS / 64, DIM / 256), 256, 0, stream>>>(
      Ao, Wprojh, b_proj, out);
}
